// EntitySelector_43112881717410
// MI455X (gfx1250) — matrix operand order repaired and checked by reading, not compiled, not run on hardware
//
#include <hip/hip_runtime.h>
#include <hip/hip_bf16.h>

typedef __attribute__((ext_vector_type(16))) _Float16 v16h;
typedef __attribute__((ext_vector_type(8)))  _Float16 v8h;
typedef __attribute__((ext_vector_type(8)))  float    v8f;
typedef __attribute__((ext_vector_type(4)))  float    v4f;

#define D_MODEL 1024
#define BATCH   16
#define LQ      1024
#define NBENT   256

// ---------------- WMMA fragment helpers (gfx1250 wave32 layouts) ----------------

__device__ __forceinline__ v8f wmma_f16(v16h a, v16h b, v8f c) {
    return __builtin_amdgcn_wmma_f32_16x16x32_f16(
        /*neg_a=*/false, a, /*neg_b=*/false, b,
        /*c_mod=*/(short)0, c, /*reuse_a=*/false, /*reuse_b=*/false);
}

// A fragment (16x32, f16 source, row-major, leading dim ld):
// lanes 0-15: M=lane, halves 0..7 -> K=kb+hi*8.., halves 8..15 -> K=kb+16+hi*8..
__device__ __forceinline__ v16h load_a_h(const _Float16* base, int ld, int kb, int lane) {
    const int m  = lane & 15;
    const int hi = lane >> 4;
    const _Float16* p = base + (size_t)m * ld + kb + hi * 8;
    v8h lo = *(const v8h*)(p);
    v8h up = *(const v8h*)(p + 16);
    v16h a;
#pragma unroll
    for (int i = 0; i < 8; ++i) { a[i] = lo[i]; a[8 + i] = up[i]; }
    return a;
}

// B fragment (32x16) for C = X @ W^T:  B(k,n) = W[n,k], W row-major (ld = K dim)
__device__ __forceinline__ v16h load_b_h(const _Float16* W, int ld, int nb, int kb, int lane) {
    const int n  = lane & 15;
    const int hi = lane >> 4;
    const _Float16* p = W + (size_t)(nb + n) * ld + kb + hi * 8;
    v8h lo = *(const v8h*)(p);
    v8h up = *(const v8h*)(p + 16);
    v16h b;
#pragma unroll
    for (int i = 0; i < 8; ++i) { b[i] = lo[i]; b[8 + i] = up[i]; }
    return b;
}

// CDNA5 async copy: 4 bytes global -> LDS, tracked by ASYNCcnt
__device__ __forceinline__ void async_copy_b32(const _Float16* gsrc, _Float16* lds_dst) {
    unsigned lds_off = (unsigned)(uintptr_t)lds_dst;   // generic shared ptr low 32 bits = LDS offset
    asm volatile("global_load_async_to_lds_b32 %0, %1, off"
                 :: "v"(lds_off), "v"(gsrc) : "memory");
}
__device__ __forceinline__ void wait_async0() {
    asm volatile("s_wait_asynccnt 0x0" ::: "memory");
}

// ---------------- kernel 0: f32 -> f16 convert ----------------

__global__ void k_cvt_f16(const float* __restrict__ src, _Float16* __restrict__ dst, int n) {
    int i = (blockIdx.x * blockDim.x + threadIdx.x) * 8;
    if (i >= n) return;
    v4f f0 = *(const v4f*)(src + i);
    v4f f1 = *(const v4f*)(src + i + 4);
    v8h h;
#pragma unroll
    for (int j = 0; j < 4; ++j) { h[j] = (_Float16)f0[j]; h[4 + j] = (_Float16)f1[j]; }
    *(v8h*)(dst + i) = h;
}

// ---------------- kernel 1: Y = A @ W^T + bias (f16 in/out), M=32 blocks ----------------
// 512 threads / 16 waves; each wave: 4 N-tiles x 2 M-tiles = 8 accumulators (64 VGPRs).
// A chunk (32x32 f16 = 2KB) staged in LDS via async b32 copies, double buffered.

__global__ void __launch_bounds__(512, 1)
k_gemm_bias_f16(const _Float16* __restrict__ A, const _Float16* __restrict__ Wh,
                const float* __restrict__ bias, _Float16* __restrict__ Yh) {
    const int row0 = blockIdx.x * 32;
    const int tid  = threadIdx.x;
    const int wave = tid >> 5;
    const int lane = tid & 31;
    const int nl = lane & 15, hi = lane >> 4;

    __shared__ _Float16 sA[2][32 * 32];

    // staging coords: 512 threads x 4B = 2KB = one 32x32 f16 chunk
    const int srow = tid >> 4;
    const int scol = (tid & 15) * 2;
    const _Float16* gsrc_base = A + (size_t)(row0 + srow) * D_MODEL + scol;

    v8f acc0[4] = {};   // M-tile 0 (rows 0-15)
    v8f acc1[4] = {};   // M-tile 1 (rows 16-31)

    async_copy_b32(gsrc_base, &sA[0][srow * 32 + scol]);
    wait_async0();
    __syncthreads();

    for (int i = 0; i < 32; ++i) {
        const int buf = i & 1;
        if (i + 1 < 32)
            async_copy_b32(gsrc_base + (i + 1) * 32, &sA[buf ^ 1][srow * 32 + scol]);

        v16h a0 = load_a_h(&sA[buf][0],       32, 0, lane);
        v16h a1 = load_a_h(&sA[buf][16 * 32], 32, 0, lane);
        const int kb = i * 32;
#pragma unroll
        for (int t = 0; t < 4; ++t) {
            v16h b = load_b_h(Wh, D_MODEL, (wave * 4 + t) * 16, kb, lane);
            acc0[t] = wmma_f16(a0, b, acc0[t]);
            acc1[t] = wmma_f16(a1, b, acc1[t]);
        }
        wait_async0();
        __syncthreads();
    }

#pragma unroll
    for (int t = 0; t < 4; ++t) {
        int n = (wave * 4 + t) * 16 + nl;
        float bn = bias[n];
#pragma unroll
        for (int r = 0; r < 8; ++r) {
            Yh[(size_t)(row0 +      r + 8 * hi) * D_MODEL + n] = (_Float16)(acc0[t][r] + bn);
            Yh[(size_t)(row0 + 16 + r + 8 * hi) * D_MODEL + n] = (_Float16)(acc1[t][r] + bn);
        }
    }
}

// ---------------- kernel 2: ent = (gather(emb,idx) @ WK^T + b) * mask ----------------
// dual store: Eh (B,NB,D) row-major and Et (B,D,NB) transposed
// grid: B*(NB/16) blocks, 256 threads

__global__ void k_proj_ent(const float* __restrict__ emb, const int* __restrict__ idx,
                           const _Float16* __restrict__ Wh, const float* __restrict__ bias,
                           _Float16* __restrict__ Eh, _Float16* __restrict__ Et) {
    const int b  = blockIdx.x >> 4;
    const int e0 = (blockIdx.x & 15) * 16;
    __shared__ int   s_idx[16];
    __shared__ float s_msk[16];
    if (threadIdx.x < 16) {
        int v = idx[b * NBENT + e0 + threadIdx.x];
        s_idx[threadIdx.x] = (v < 0) ? 0 : v;        // value irrelevant when masked
        s_msk[threadIdx.x] = (v < 0) ? 0.0f : 1.0f;
    }
    __syncthreads();
    const int wave = threadIdx.x >> 5;
    const int lane = threadIdx.x & 31;
    const int m = lane & 15, hi = lane >> 4;
    const float* arow = emb + (size_t)s_idx[m] * D_MODEL + hi * 8;
    v8f acc[8] = {};
    for (int kb = 0; kb < D_MODEL; kb += 32) {
        const float* p = arow + kb;
        v4f f0 = *(const v4f*)(p);
        v4f f1 = *(const v4f*)(p + 4);
        v4f f2 = *(const v4f*)(p + 16);
        v4f f3 = *(const v4f*)(p + 20);
        v16h a;
#pragma unroll
        for (int i = 0; i < 4; ++i) {
            a[i] = (_Float16)f0[i]; a[4 + i] = (_Float16)f1[i];
            a[8 + i] = (_Float16)f2[i]; a[12 + i] = (_Float16)f3[i];
        }
#pragma unroll
        for (int t = 0; t < 8; ++t)
            acc[t] = wmma_f16(a, load_b_h(Wh, D_MODEL, (wave * 8 + t) * 16, kb, lane), acc[t]);
    }
#pragma unroll
    for (int t = 0; t < 8; ++t) {
        int n = (wave * 8 + t) * 16 + m;
        float bn = bias[n];
        v8h pk;
#pragma unroll
        for (int r = 0; r < 8; ++r) {
            float x = (acc[t][r] + bn) * s_msk[r + 8 * hi];
            pk[r] = (_Float16)x;
            Eh[(size_t)b * NBENT * D_MODEL + (size_t)(e0 + r + 8 * hi) * D_MODEL + n] = pk[r];
        }
        *(v8h*)(Et + (size_t)b * D_MODEL * NBENT + (size_t)n * NBENT + e0 + 8 * hi) = pk;
    }
}

// ---------------- kernel 3: scores + softmax + weighted sum (fused attention) ----------------
// grid: B*(L/16) blocks, 256 threads

__global__ void k_attn(const _Float16* __restrict__ Qh, const _Float16* __restrict__ Eh,
                       const _Float16* __restrict__ Et, const int* __restrict__ idx,
                       _Float16* __restrict__ Vh) {
    const int b    = blockIdx.x >> 6;
    const int row0 = (blockIdx.x & 63) * 16;
    const int wave = threadIdx.x >> 5;
    const int lane = threadIdx.x & 31;
    const int nl = lane & 15, hi = lane >> 4;

    __shared__ float    sS[16][NBENT];   // raw scores
    __shared__ _Float16 sP[16][NBENT];   // probabilities (f16, WMMA-A source)

    const _Float16* qbase = Qh + ((size_t)b * LQ + row0) * D_MODEL;
    const _Float16* ebase = Eh + (size_t)b * NBENT * D_MODEL;

    // GEMM1: S = q_tile @ ent^T (16 x 256, K=1024); each wave: 2 N-tiles
    v8f s0 = {}, s1 = {};
    for (int kb = 0; kb < D_MODEL; kb += 32) {
        v16h a = load_a_h(qbase, D_MODEL, kb, lane);
        s0 = wmma_f16(a, load_b_h(ebase, D_MODEL, wave * 32,      kb, lane), s0);
        s1 = wmma_f16(a, load_b_h(ebase, D_MODEL, wave * 32 + 16, kb, lane), s1);
    }
#pragma unroll
    for (int r = 0; r < 8; ++r) {
        sS[r + 8 * hi][wave * 32 + nl]      = s0[r];
        sS[r + 8 * hi][wave * 32 + 16 + nl] = s1[r];
    }
    __syncthreads();

    // masked stable softmax; reference scales by D^-0.5 = 1/32 AFTER softmax
#pragma unroll
    for (int rr = 0; rr < 2; ++rr) {
        const int row = wave * 2 + rr;
        float vals[8];
        float mx = -3.0e38f;
#pragma unroll
        for (int j = 0; j < 8; ++j) {
            int c = lane * 8 + j;
            float s = sS[row][c];
            if (idx[b * NBENT + c] < 0) s -= 100000.0f;
            vals[j] = s;
            mx = fmaxf(mx, s);
        }
#pragma unroll
        for (int off = 16; off > 0; off >>= 1) mx = fmaxf(mx, __shfl_xor(mx, off, 32));
        float sum = 0.0f;
#pragma unroll
        for (int j = 0; j < 8; ++j) { vals[j] = __expf(vals[j] - mx); sum += vals[j]; }
#pragma unroll
        for (int off = 16; off > 0; off >>= 1) sum += __shfl_xor(sum, off, 32);
        const float inv = 0.03125f / sum;
#pragma unroll
        for (int j = 0; j < 8; ++j) sP[row][lane * 8 + j] = (_Float16)(vals[j] * inv);
    }
    __syncthreads();

    // GEMM2: v = P(16x256) @ ent(256x1024); B from transposed Et -> contiguous loads
    const _Float16* tbase = Et + (size_t)b * D_MODEL * NBENT;
    v8f vacc[8] = {};
    for (int kb = 0; kb < NBENT; kb += 32) {
        v16h a = load_a_h(&sP[0][0], NBENT, kb, lane);
#pragma unroll
        for (int t = 0; t < 8; ++t)
            vacc[t] = wmma_f16(a, load_b_h(tbase, NBENT, (wave * 8 + t) * 16, kb, lane), vacc[t]);
    }
#pragma unroll
    for (int t = 0; t < 8; ++t) {
        int n = (wave * 8 + t) * 16 + nl;
#pragma unroll
        for (int r = 0; r < 8; ++r)
            Vh[((size_t)b * LQ + row0 + r + 8 * hi) * D_MODEL + n] = (_Float16)vacc[t][r];
    }
}

// ---------------- kernel 4: out = LN(v @ WO^T + b), M=32, async-staged A ----------------
// 512 threads / 16 waves; each wave: 4 N-tiles x 2 M-tiles

__global__ void __launch_bounds__(512, 1)
k_out_ln(const _Float16* __restrict__ Vh, const _Float16* __restrict__ Wh,
         const float* __restrict__ bias, const float* __restrict__ g,
         const float* __restrict__ be, float* __restrict__ out) {
    const int row0 = blockIdx.x * 32;
    const int tid  = threadIdx.x;
    const int wave = tid >> 5;
    const int lane = tid & 31;
    const int nl = lane & 15, hi = lane >> 4;

    __shared__ _Float16 sA[2][32 * 32];
    __shared__ float sSum[16][32];
    __shared__ float sSq[16][32];
    __shared__ float sMu[32];
    __shared__ float sRs[32];

    const int srow = tid >> 4;
    const int scol = (tid & 15) * 2;
    const _Float16* gsrc_base = Vh + (size_t)(row0 + srow) * D_MODEL + scol;

    v8f acc0[4] = {};
    v8f acc1[4] = {};

    async_copy_b32(gsrc_base, &sA[0][srow * 32 + scol]);
    wait_async0();
    __syncthreads();

    for (int i = 0; i < 32; ++i) {
        const int buf = i & 1;
        if (i + 1 < 32)
            async_copy_b32(gsrc_base + (i + 1) * 32, &sA[buf ^ 1][srow * 32 + scol]);

        v16h a0 = load_a_h(&sA[buf][0],       32, 0, lane);
        v16h a1 = load_a_h(&sA[buf][16 * 32], 32, 0, lane);
        const int kb = i * 32;
#pragma unroll
        for (int t = 0; t < 4; ++t) {
            v16h b = load_b_h(Wh, D_MODEL, (wave * 4 + t) * 16, kb, lane);
            acc0[t] = wmma_f16(a0, b, acc0[t]);
            acc1[t] = wmma_f16(a1, b, acc1[t]);
        }
        wait_async0();
        __syncthreads();
    }

    // bias add + per-row partial moments (this wave covers 64 of 1024 columns)
    float ps0[8] = {}, pq0[8] = {}, ps1[8] = {}, pq1[8] = {};
#pragma unroll
    for (int t = 0; t < 4; ++t) {
        float bn = bias[(wave * 4 + t) * 16 + nl];
#pragma unroll
        for (int r = 0; r < 8; ++r) {
            float x0 = acc0[t][r] + bn;  acc0[t][r] = x0;  ps0[r] += x0;  pq0[r] += x0 * x0;
            float x1 = acc1[t][r] + bn;  acc1[t][r] = x1;  ps1[r] += x1;  pq1[r] += x1 * x1;
        }
    }
#pragma unroll
    for (int off = 8; off > 0; off >>= 1) {
#pragma unroll
        for (int r = 0; r < 8; ++r) {
            ps0[r] += __shfl_xor(ps0[r], off, 32);  pq0[r] += __shfl_xor(pq0[r], off, 32);
            ps1[r] += __shfl_xor(ps1[r], off, 32);  pq1[r] += __shfl_xor(pq1[r], off, 32);
        }
    }
    if (nl == 0) {
#pragma unroll
        for (int r = 0; r < 8; ++r) {
            sSum[wave][     r + 8 * hi] = ps0[r];  sSq[wave][     r + 8 * hi] = pq0[r];
            sSum[wave][16 + r + 8 * hi] = ps1[r];  sSq[wave][16 + r + 8 * hi] = pq1[r];
        }
    }
    __syncthreads();
    if (tid < 32) {
        float s = 0.0f, q = 0.0f;
#pragma unroll
        for (int w = 0; w < 16; ++w) { s += sSum[w][tid]; q += sSq[w][tid]; }
        float mu  = s * (1.0f / D_MODEL);
        float var = q * (1.0f / D_MODEL) - mu * mu;
        sMu[tid] = mu;
        sRs[tid] = rsqrtf(var + 1e-5f);
    }
    __syncthreads();
#pragma unroll
    for (int t = 0; t < 4; ++t) {
        int n = (wave * 4 + t) * 16 + nl;
        float gn = g[n], bn = be[n];
#pragma unroll
        for (int r = 0; r < 8; ++r) {
            int m0 = r + 8 * hi;
            int m1 = 16 + r + 8 * hi;
            out[(size_t)(row0 + m0) * D_MODEL + n] = (acc0[t][r] - sMu[m0]) * sRs[m0] * gn + bn;
            out[(size_t)(row0 + m1) * D_MODEL + n] = (acc1[t][r] - sMu[m1]) * sRs[m1] * gn + bn;
        }
    }
}

// ---------------- host launcher ----------------

extern "C" void kernel_launch(void* const* d_in, const int* in_sizes, int n_in,
                              void* d_out, int out_size, void* d_ws, size_t ws_size,
                              hipStream_t stream) {
    const float* query   = (const float*)d_in[0];
    const float* ent_emb = (const float*)d_in[1];
    const int*   idx     = (const int*)d_in[2];
    const float* WQ_w    = (const float*)d_in[4];
    const float* WQ_b    = (const float*)d_in[5];
    const float* WK_w    = (const float*)d_in[6];
    const float* WK_b    = (const float*)d_in[7];
    const float* WO_w    = (const float*)d_in[8];
    const float* WO_b    = (const float*)d_in[9];
    const float* ln_g    = (const float*)d_in[10];
    const float* ln_b    = (const float*)d_in[11];
    float* out = (float*)d_out;

    // workspace layout (f16 intermediates); Vh aliases Xh (query_h consumed by q-proj
    // before k_attn writes Vh; stream-ordered so this is safe)
    char* ws = (char*)d_ws;
    _Float16* WQh = (_Float16*)ws;  ws += (size_t)D_MODEL * D_MODEL * 2;
    _Float16* WKh = (_Float16*)ws;  ws += (size_t)D_MODEL * D_MODEL * 2;
    _Float16* WOh = (_Float16*)ws;  ws += (size_t)D_MODEL * D_MODEL * 2;
    _Float16* Xh  = (_Float16*)ws;  ws += (size_t)BATCH * LQ * D_MODEL * 2;   // query f16, reused as Vh
    _Float16* Qh  = (_Float16*)ws;  ws += (size_t)BATCH * LQ * D_MODEL * 2;
    _Float16* Eh  = (_Float16*)ws;  ws += (size_t)BATCH * NBENT * D_MODEL * 2;
    _Float16* Et  = (_Float16*)ws;
    _Float16* Vh  = Xh;

    const int nW = D_MODEL * D_MODEL;
    const int nQ = BATCH * LQ * D_MODEL;
    k_cvt_f16<<<nW / 2048, 256, 0, stream>>>(WQ_w, WQh, nW);
    k_cvt_f16<<<nW / 2048, 256, 0, stream>>>(WK_w, WKh, nW);
    k_cvt_f16<<<nW / 2048, 256, 0, stream>>>(WO_w, WOh, nW);
    k_cvt_f16<<<nQ / 2048, 256, 0, stream>>>(query, Xh, nQ);

    k_gemm_bias_f16<<<(BATCH * LQ) / 32, 512, 0, stream>>>(Xh, WQh, WQ_b, Qh);
    k_proj_ent<<<BATCH * (NBENT / 16), 256, 0, stream>>>(ent_emb, idx, WKh, WK_b, Eh, Et);
    k_attn    <<<BATCH * (LQ / 16), 256, 0, stream>>>(Qh, Eh, Et, idx, Vh);
    k_out_ln  <<<(BATCH * LQ) / 32, 512, 0, stream>>>(Vh, WOh, WO_b, ln_g, ln_b, out);
}
